// Encoder_13297218748546
// MI455X (gfx1250) — matrix operand order repaired and checked by reading, not compiled, not run on hardware
//
#include <hip/hip_runtime.h>
#include <hip/hip_bf16.h>

// ---------------------------------------------------------------------------
// GRU encoder + windowed segment scans for MI455X (gfx1250, wave32).
// All activations are kept in WMMA A-fragment layout (f16, K padded to 32),
// so every recurrence step is: 2x b128 load (x frag) + 2x b128 LDS load
// (h frag) + 4x v_wmma_f32_16x16x32_f16 (bias folded into C) + LDS bounce of
// the D tiles + branch-free elementwise update. Hidden state lives in
// registers (5 elements per lane).
// ---------------------------------------------------------------------------

typedef __attribute__((ext_vector_type(16))) _Float16 v16h;
typedef __attribute__((ext_vector_type(8)))  float    v8f;
typedef unsigned short ushort_t;

#define NSEQ   256
#define BATCH  16
#define SDIM   10
#define LNUM   6
#define TWIN   5
#define NPAIR  32896                                  // N*(N+1)/2
#define DPSZ   ((size_t)NPAIR * BATCH * 2 * SDIM)     // 10,526,720 floats

struct GruP {
    const float* Wih;   // (30, in_dim) row-major
    const float* Whh;   // (30, 10)     row-major
    const float* bih;   // (30)
    const float* bhh;   // (30)
    int in_dim;
};
struct AllParams {
    GruP enc[6][2];     // [layer][0=f,1=b]
    GruP seg[6];
};

// ---------------------------------------------------------------------------
// Fragment layout helpers (CDNA5 ISA §7.12.2)
// A 16x32 f16:  lane<16: M=lane,    e -> K = e (e<8), e+8  (e>=8)
//               lane>=16:M=lane-16, e -> K = e+8 (e<8), e+16 (e>=8)
// value (m, k) lives at lane' = m + ((k&8)?16:0), e = (k&7) + ((k&16)?8:0)
// ---------------------------------------------------------------------------
__device__ __forceinline__ int act_lane(int m, int k) { return m + ((k & 8) ? 16 : 0); }
__device__ __forceinline__ int act_elem(int k)        { return (k & 7) | ((k & 16) >> 1); }

__device__ __forceinline__ v16h load_v16h(const _Float16* p) { return *(const v16h*)p; }

__device__ __forceinline__ v16h load_frag(const _Float16* fragW, int f, int lane) {
    return *(const v16h*)(fragW + ((size_t)f * 32 + lane) * 16);
}

__device__ __forceinline__ v8f splat8(float v) {
    v8f r;
#pragma unroll
    for (int i = 0; i < 8; ++i) r[i] = v;
    return r;
}

// 32-bit D 16x16: VGPR r holds (M = r + 8*(lane>=16), N = lane&15)
__device__ __forceinline__ void store_d(float* dst, v8f d, int lane, int ncol_off) {
    const int n     = (lane & 15) + ncol_off;
    const int mbase = (lane & 16) ? 8 : 0;
#pragma unroll
    for (int r = 0; r < 8; ++r) dst[(mbase + r) * 32 + n] = d[r];
}

__device__ __forceinline__ float sigm_f(float x) { return 1.0f / (1.0f + __expf(-x)); }
__device__ __forceinline__ float tanh_f(float x) { return 1.0f - 2.0f / (__expf(2.0f * x) + 1.0f); }

// ---------------------------------------------------------------------------
// One GRU cell step. Biases are pre-folded into the C fragments.
//   xfl : this lane's 32B slice of A(x) fragment (global or LDS)
//   hfl : this lane's 32B slice of A(h) fragment (LDS)
//   hfrag: base of the h fragment (LDS) for the scatter update
//   hreg : this lane's 5 state elements (b = lane>>1, s = s0..s0+4)
//   yout : optional fragment-layout output scatter (encoder), keys ky0..ky0+9
//   dpdst: optional f32 output (segment last layer)
// ---------------------------------------------------------------------------
__device__ __forceinline__ void gru_cell(
    const _Float16* xfl, const _Float16* hfl, _Float16* hfrag,
    v16h w0, v16h w1, v16h w2, v16h w3,
    v8f c0, v8f c1, v8f c2, v8f c3,
    float* gi, float* gh, float hreg[5],
    _Float16* yout, int ky0, float* dpdst, int lane)
{
    v16h ax = load_v16h(xfl);
    v8f d0 = __builtin_amdgcn_wmma_f32_16x16x32_f16(false, ax, false, w0, (short)0, c0, false, false);
    v8f d1 = __builtin_amdgcn_wmma_f32_16x16x32_f16(false, ax, false, w1, (short)0, c1, false, false);
    store_d(gi, d0, lane, 0);
    store_d(gi, d1, lane, 16);

    v16h ah = load_v16h(hfl);
    v8f e0 = __builtin_amdgcn_wmma_f32_16x16x32_f16(false, ah, false, w2, (short)0, c2, false, false);
    v8f e1 = __builtin_amdgcn_wmma_f32_16x16x32_f16(false, ah, false, w3, (short)0, c3, false, false);
    store_d(gh, e0, lane, 0);
    store_d(gh, e1, lane, 16);

    asm volatile("s_wait_dscnt 0x0" ::: "memory");   // D tiles visible wave-wide

    const int b  = lane >> 1;
    const int s0 = (lane & 1) * 5;
#pragma unroll
    for (int q = 0; q < 5; ++q) {
        int s = s0 + q;
        float ir  = gi[b * 32 + s];
        float iz  = gi[b * 32 + s + 10];
        float inn = gi[b * 32 + s + 20];
        float hr  = gh[b * 32 + s];
        float hz  = gh[b * 32 + s + 10];
        float hn  = gh[b * 32 + s + 20];
        float r  = sigm_f(ir + hr);
        float z  = sigm_f(iz + hz);
        float nn = tanh_f(inn + r * hn);
        float hv = (1.0f - z) * nn + z * hreg[q];
        hreg[q] = hv;
        hfrag[act_lane(b, s) * 16 + act_elem(s)] = (_Float16)hv;   // k = s < 10
        if (yout) {
            int k = ky0 + s;
            yout[act_lane(b, k) * 16 + act_elem(k)] = (_Float16)hv;
        }
        if (dpdst) dpdst[b * 20 + s] = hv;
    }
    asm volatile("s_wait_dscnt 0x0" ::: "memory");   // h fragment visible wave-wide
}

// ---------------------------------------------------------------------------
// Pack weights into per-lane B-fragment layout (f16).
// B 32x16 f16: lane<16 col N=lane, K=e (e<8), e+8 (e>=8); lane>=16 K=e+8 (e<8), e+16 (e>=8).  B[k][n] = W[n][k].
// enc cell (l*2+dir): frags cell*4+{0..3}; seg layer l: 48+l*4.
// ---------------------------------------------------------------------------
__global__ void prep_frags(ushort_t* __restrict__ fragW_u, AllParams P) {
    _Float16* fb = reinterpret_cast<_Float16*>(fragW_u);
    int f = blockIdx.x;          // 0..71
    int lane = threadIdx.x;      // 0..31
    const float* W; int in_dim, which, tile;
    if (f < 48) {
        int cell = f >> 2; int l = cell >> 1, d = cell & 1;
        which = (f >> 1) & 1; tile = f & 1;
        W = which ? P.enc[l][d].Whh : P.enc[l][d].Wih;
        in_dim = which ? SDIM : P.enc[l][d].in_dim;
    } else {
        int ff = f - 48; int l = ff >> 2;
        which = (ff >> 1) & 1; tile = ff & 1;
        W = which ? P.seg[l].Whh : P.seg[l].Wih;
        in_dim = which ? SDIM : P.seg[l].in_dim;
    }
    int n  = (lane & 15) + tile * 16;
    int kb = (lane & 16) ? 8 : 0;
    _Float16* dst = fb + ((size_t)f * 32 + lane) * 16;
#pragma unroll
    for (int e = 0; e < 16; ++e) {
        int k = (e & 7) + kb + ((e >= 8) ? 16 : 0);
        float v = (n < 30 && k < in_dim) ? W[(size_t)n * in_dim + k] : 0.0f;
        dst[e] = (_Float16)v;
    }
}

// ---------------------------------------------------------------------------
// Embedding straight into fragment layout: one thread per (t, lane),
// writes its full 32B slice (pad elements zeroed, loads clamped -> no branch).
// ---------------------------------------------------------------------------
__global__ void embed_kernel(const int* __restrict__ tok,
                             const float* __restrict__ emb,
                             ushort_t* __restrict__ xemb_u) {
    int gid = blockIdx.x * blockDim.x + threadIdx.x;     // 256*32
    if (gid >= NSEQ * 32) return;
    int t = gid >> 5, lane = gid & 31;
    int m = lane & 15, kb = (lane & 16) ? 8 : 0;
    const float* er = emb + (size_t)tok[t * BATCH + m] * 10;
    v16h a;
#pragma unroll
    for (int e = 0; e < 16; ++e) {
        int k = (e & 7) + kb + ((e >= 8) ? 16 : 0);
        float v = er[min(k, 9)];
        a[e] = (_Float16)((k < 10) ? v : 0.0f);
    }
    *(v16h*)(reinterpret_cast<_Float16*>(xemb_u) + (size_t)gid * 16) = a;
}

__global__ void zero_kernel(float* __restrict__ p, long n) {
    long i = (long)blockIdx.x * blockDim.x + threadIdx.x;
    long stride = (long)gridDim.x * blockDim.x;
    for (; i < n; i += stride) p[i] = 0.0f;
}

// ---------------------------------------------------------------------------
// Encoder: 1 block, 2 waves (wave0 fwd, wave1 bwd), layers sequential.
// Weight fragments live in registers across the 256-step recurrence.
// Layer outputs ping-pong between fragment-layout f16 buffers.
// ---------------------------------------------------------------------------
__global__ void __launch_bounds__(64) encoder_kernel(
    const ushort_t* __restrict__ xemb_u, ushort_t* __restrict__ buf0_u,
    ushort_t* __restrict__ buf1_u, const ushort_t* __restrict__ fragW_u,
    AllParams P)
{
    const _Float16* xemb  = reinterpret_cast<const _Float16*>(xemb_u);
    _Float16* buf0        = reinterpret_cast<_Float16*>(buf0_u);
    _Float16* buf1        = reinterpret_cast<_Float16*>(buf1_u);
    const _Float16* fragW = reinterpret_cast<const _Float16*>(fragW_u);
    const int lane = threadIdx.x & 31;
    const int dir  = threadIdx.x >> 5;          // 0 = fwd, 1 = bwd

    __shared__ __align__(32) _Float16 hfrag[2][32 * 16];
    __shared__ float gi[2][BATCH * 32];
    __shared__ float gh[2][BATCH * 32];

    float hreg[5];
    const int nn = lane & 15;

    for (int l = 0; l < LNUM; ++l) {
        const _Float16* in = (l == 0) ? xemb : ((l & 1) ? buf0 : buf1);
        _Float16* out      = (l & 1) ? buf1 : buf0;
        const int fbase    = (l * 2 + dir) * 4;
        const GruP& g      = P.enc[l][dir];

        v16h w0 = load_frag(fragW, fbase + 0, lane);
        v16h w1 = load_frag(fragW, fbase + 1, lane);
        v16h w2 = load_frag(fragW, fbase + 2, lane);
        v16h w3 = load_frag(fragW, fbase + 3, lane);

        // bias C fragments (clamped load + select -> branchless)
        float bi1r = g.bih[min(nn + 16, 29)];
        float bh1r = g.bhh[min(nn + 16, 29)];
        v8f c0 = splat8(g.bih[nn]);
        v8f c1 = splat8((nn + 16 < 30) ? bi1r : 0.0f);
        v8f c2 = splat8(g.bhh[nn]);
        v8f c3 = splat8((nn + 16 < 30) ? bh1r : 0.0f);

#pragma unroll
        for (int q = 0; q < 5; ++q) hreg[q] = 0.0f;
        { v16h zz = {}; *(v16h*)(hfrag[dir] + lane * 16) = zz; }
        asm volatile("s_wait_dscnt 0x0" ::: "memory");

        for (int step = 0; step < NSEQ; ++step) {
            int t = dir ? (NSEQ - 1 - step) : step;
            gru_cell(in + ((size_t)t * 32 + lane) * 16,
                     hfrag[dir] + lane * 16, hfrag[dir],
                     w0, w1, w2, w3, c0, c1, c2, c3,
                     gi[dir], gh[dir], hreg,
                     out + (size_t)t * 32 * 16, dir * 10, nullptr, lane);
        }
        __threadfence();        // layer output visible to the other wave
        __syncthreads();
    }
}

// ---------------------------------------------------------------------------
// Segment scans: 512 independent (dir, i) scans, one wave each.
// Layer loop fully unrolled so per-layer state stays in registers.
// ---------------------------------------------------------------------------
__global__ void __launch_bounds__(128) segment_kernel(
    const ushort_t* __restrict__ outs_u,     // buf1 fragment-layout (N,32,16)
    const float* __restrict__ h0f, const float* __restrict__ h0b,
    const ushort_t* __restrict__ fragW_u, AllParams P,
    float* __restrict__ dout)
{
    const _Float16* outs  = reinterpret_cast<const _Float16*>(outs_u);
    const _Float16* fragW = reinterpret_cast<const _Float16*>(fragW_u);
    const int lane = threadIdx.x & 31;
    const int w    = threadIdx.x >> 5;
    const int wid  = blockIdx.x * 4 + w;       // 0..511
    const int dir  = wid >> 8;                 // 0 = fwd, 1 = bwd
    const int i    = wid & 255;

    __shared__ __align__(32) _Float16 hfrag[4][LNUM][32 * 16];
    __shared__ float gi[4][BATCH * 32];
    __shared__ float gh[4][BATCH * 32];

    const float* h0 = (dir ? h0b : h0f) + (size_t)i * (LNUM * BATCH * SDIM);
    const int b  = lane >> 1;
    const int s0 = (lane & 1) * 5;
    const int m  = lane & 15;
    const int kb = (lane & 16) ? 8 : 0;

    float hreg[LNUM][5];
    float bi0[LNUM], bi1[LNUM], bh0[LNUM], bh1[LNUM];
    const int nn = lane & 15;

#pragma unroll
    for (int l = 0; l < LNUM; ++l) {
        // register state
#pragma unroll
        for (int q = 0; q < 5; ++q) hreg[l][q] = h0[l * 160 + b * 10 + s0 + q];
        // f16 fragment copy of h (clamped loads + select, no branches)
        v16h hv;
#pragma unroll
        for (int e = 0; e < 16; ++e) {
            int k = (e & 7) + kb + ((e >= 8) ? 16 : 0);
            float v = h0[l * 160 + m * 10 + min(k, 9)];
            hv[e] = (_Float16)((k < 10) ? v : 0.0f);
        }
        *(v16h*)(hfrag[w][l] + lane * 16) = hv;
        // bias scalars
        const GruP& g = P.seg[l];
        float bi1r = g.bih[min(nn + 16, 29)];
        float bh1r = g.bhh[min(nn + 16, 29)];
        bi0[l] = g.bih[nn];
        bi1[l] = (nn + 16 < 30) ? bi1r : 0.0f;
        bh0[l] = g.bhh[nn];
        bh1[l] = (nn + 16 < 30) ? bh1r : 0.0f;
    }
    asm volatile("s_wait_dscnt 0x0" ::: "memory");

    const int nT = dir ? min(TWIN, i + 1) : min(TWIN, NSEQ - i);

    for (int t = 0; t < nT; ++t) {
        const int src = dir ? (i - t) : (i + t);
        const int r   = dir ? (i - t) : i;
        const int c   = dir ? i : (i + t);
        const size_t p = (size_t)r * NSEQ - (size_t)r * (r - 1) / 2 + (size_t)(c - r);
        float* dst = dout + p * (BATCH * 2 * SDIM) + dir * SDIM;

#pragma unroll
        for (int l = 0; l < LNUM; ++l) {
            const _Float16* xfl = (l == 0)
                ? (outs + ((size_t)src * 32 + lane) * 16)
                : (hfrag[w][l - 1] + lane * 16);
            v16h w0 = load_frag(fragW, 48 + l * 4 + 0, lane);
            v16h w1 = load_frag(fragW, 48 + l * 4 + 1, lane);
            v16h w2 = load_frag(fragW, 48 + l * 4 + 2, lane);
            v16h w3 = load_frag(fragW, 48 + l * 4 + 3, lane);
            gru_cell(xfl, hfrag[w][l] + lane * 16, hfrag[w][l],
                     w0, w1, w2, w3,
                     splat8(bi0[l]), splat8(bi1[l]), splat8(bh0[l]), splat8(bh1[l]),
                     gi[w], gh[w], hreg[l],
                     nullptr, 0, (l == LNUM - 1) ? dst : nullptr, lane);
        }
    }

    if (i == NSEQ - 1) {   // hidden = concat(hT_f[255], hT_b[255]) -> (2L,B,S)
        float* hid = dout + DPSZ + (size_t)dir * LNUM * BATCH * SDIM;
#pragma unroll
        for (int l = 0; l < LNUM; ++l)
#pragma unroll
            for (int q = 0; q < 5; ++q)
                hid[l * 160 + b * 10 + s0 + q] = hreg[l][q];
    }
}

// ---------------------------------------------------------------------------
extern "C" void kernel_launch(void* const* d_in, const int* in_sizes, int n_in,
                              void* d_out, int out_size, void* d_ws, size_t ws_size,
                              hipStream_t stream) {
    (void)in_sizes; (void)n_in; (void)ws_size;
    const int*   tokens = (const int*)d_in[0];
    const float* emb    = (const float*)d_in[1];

    AllParams P;
    for (int l = 0; l < 6; ++l)
        for (int d = 0; d < 2; ++d) {
            int base = 2 + (l * 2 + d) * 4;          // f then b per layer
            P.enc[l][d].Wih = (const float*)d_in[base + 0];
            P.enc[l][d].Whh = (const float*)d_in[base + 1];
            P.enc[l][d].bih = (const float*)d_in[base + 2];
            P.enc[l][d].bhh = (const float*)d_in[base + 3];
            P.enc[l][d].in_dim = (l == 0) ? 10 : 20;
        }
    for (int l = 0; l < 6; ++l) {
        int base = 50 + l * 4;
        P.seg[l].Wih = (const float*)d_in[base + 0];
        P.seg[l].Whh = (const float*)d_in[base + 1];
        P.seg[l].bih = (const float*)d_in[base + 2];
        P.seg[l].bhh = (const float*)d_in[base + 3];
        P.seg[l].in_dim = (l == 0) ? 20 : 10;
    }
    const float* h0f = (const float*)d_in[74];
    const float* h0b = (const float*)d_in[75];

    // workspace layout (bytes, 256B-aligned sections)
    char* ws = (char*)d_ws;
    ushort_t* fragW = (ushort_t*)(ws);                    // 72*32*16*2  =  73,728 B
    ushort_t* xemb  = (ushort_t*)(ws + 73728);            // 256*32*16*2 = 262,144 B
    ushort_t* buf0  = (ushort_t*)(ws + 73728 + 262144);   // 262,144 B
    ushort_t* buf1  = (ushort_t*)(ws + 73728 + 2 * 262144);
    float* dout = (float*)d_out;

    prep_frags  <<<72, 32, 0, stream>>>(fragW, P);
    embed_kernel<<<(NSEQ * 32 + 255) / 256, 256, 0, stream>>>(tokens, emb, xemb);
    zero_kernel <<<2048, 256, 0, stream>>>(dout, (long)out_size);
    zero_kernel <<<256, 256, 0, stream>>>((float*)buf0, (long)(2 * 262144 / 4));
    encoder_kernel<<<1, 64, 0, stream>>>(xemb, buf0, buf1, fragW, P);
    segment_kernel<<<128, 128, 0, stream>>>(buf1, h0f, h0b, fragW, P, dout);
}
